// DecoderRefineHead_70927089926649
// MI455X (gfx1250) — compile-verified, matrix operand order checked by reading, not run on hardware
//
#include <hip/hip_runtime.h>
#include <math.h>

// ---------------------------------------------------------------------------
// CDNA5 (gfx1250) fused point-refinement head.
// All dense layers run on v_wmma_f32_16x16x32_f16 (wave32, M-tile=16/wave).
// ---------------------------------------------------------------------------

typedef __attribute__((ext_vector_type(16))) _Float16 v16h;
typedef __attribute__((ext_vector_type(8)))  float    v8f;

union Frag {
  uint4    q[2];
  v16h     h;
  _Float16 e[16];
};

__device__ __forceinline__ float gelu_exact(float x) {
  return 0.5f * x * (1.0f + erff(x * 0.70710678118654752f));
}
__device__ __forceinline__ float sigmoidf_(float x) {
  return 1.0f / (1.0f + expf(-x));
}
__device__ __forceinline__ float wred_sum(float v) {
#pragma unroll
  for (int o = 16; o > 0; o >>= 1) v += __shfl_xor(v, o, 32);
  return v;
}
__device__ __forceinline__ float wred_max(float v) {
#pragma unroll
  for (int o = 16; o > 0; o >>= 1) v = fmaxf(v, __shfl_xor(v, o, 32));
  return v;
}

// A fragment for V_WMMA_F32_16X16X32_F16 from a row-major f16 [rows x 128]
// array. Lane m (m<16) holds row m: K k0..k0+7 in v0..3 and k0+16..23 in
// v4..7; lane m+16 holds K k0+8..15 and k0+24..31  (ISA 7.12.2 layout).
__device__ __forceinline__ Frag load_row_frag(const _Float16* __restrict__ base,
                                              int row, int k0, int lane) {
  const _Float16* pp = base + (size_t)row * 128 + k0 + ((lane >= 16) ? 8 : 0);
  Frag f;
  f.q[0] = *(const uint4*)pp;
  f.q[1] = *(const uint4*)(pp + 16);
  return f;
}

// B fragment from pre-packed weights: per (kt,nt) tile, 32 lanes x 16 f16
// contiguous (32 B per lane, fully coalesced).
__device__ __forceinline__ Frag load_b_frag(const _Float16* __restrict__ pk,
                                            int kt, int nt, int lane) {
  const _Float16* pp = pk + ((size_t)((kt * 8 + nt) * 32 + lane) << 4);
  Frag f;
  f.q[0] = *(const uint4*)pp;
  f.q[1] = *(const uint4*)(pp + 8);
  return f;
}

// A fragment from per-wave LDS hidden buffer, row-major stride 136 f16.
__device__ __forceinline__ Frag lds_frag(const _Float16* sm, int kt, int lane) {
  const _Float16* pp = sm + (lane & 15) * 136 + kt * 32 + ((lane >= 16) ? 8 : 0);
  Frag f;
  f.q[0] = *(const uint4*)pp;
  f.q[1] = *(const uint4*)(pp + 16);
  return f;
}

__device__ __forceinline__ v8f wmma16(const Frag& a, const Frag& b, v8f c) {
  return __builtin_amdgcn_wmma_f32_16x16x32_f16(false, a.h, false, b.h,
                                                (short)0, c, false, false);
}

// ---------------------------------------------------------------------------
// Stats: deterministic two-pass mean, atomicMax (order-invariant) for scale.
// ---------------------------------------------------------------------------
#define RED_BLOCKS 256

__global__ void k_stats_init(unsigned* maxb) {
  if (threadIdx.x < 3) maxb[threadIdx.x] = 0u;
}

__global__ void k_mean_part(const float* __restrict__ pts, int n,
                            float* __restrict__ partial) {
  __shared__ float sw[8][3];
  float sx = 0.f, sy = 0.f, sz = 0.f;
  for (int i = blockIdx.x * blockDim.x + threadIdx.x; i < n;
       i += gridDim.x * blockDim.x) {
    sx += pts[(size_t)i * 3 + 0];
    sy += pts[(size_t)i * 3 + 1];
    sz += pts[(size_t)i * 3 + 2];
  }
  sx = wred_sum(sx); sy = wred_sum(sy); sz = wred_sum(sz);
  int lane = threadIdx.x & 31, wv = threadIdx.x >> 5;
  if (lane == 0) { sw[wv][0] = sx; sw[wv][1] = sy; sw[wv][2] = sz; }
  __syncthreads();
  if (threadIdx.x == 0) {
    float a = 0.f, b = 0.f, c = 0.f;
    for (int w = 0; w < 8; ++w) { a += sw[w][0]; b += sw[w][1]; c += sw[w][2]; }
    partial[blockIdx.x * 3 + 0] = a;
    partial[blockIdx.x * 3 + 1] = b;
    partial[blockIdx.x * 3 + 2] = c;
  }
}

__global__ void k_mean_fin(const float* __restrict__ partial, int n,
                           float* __restrict__ stats) {
  if (threadIdx.x == 0) {
    for (int d = 0; d < 3; ++d) {
      float s = 0.f;
      for (int b = 0; b < RED_BLOCKS; ++b) s += partial[b * 3 + d];
      stats[d] = s / (float)n;
    }
  }
}

__global__ void k_maxabs(const float* __restrict__ pts, int n,
                         const float* __restrict__ stats, unsigned* maxb) {
  float m0 = stats[0], m1 = stats[1], m2 = stats[2];
  float a0 = 0.f, a1 = 0.f, a2 = 0.f;
  for (int i = blockIdx.x * blockDim.x + threadIdx.x; i < n;
       i += gridDim.x * blockDim.x) {
    a0 = fmaxf(a0, fabsf(pts[(size_t)i * 3 + 0] - m0));
    a1 = fmaxf(a1, fabsf(pts[(size_t)i * 3 + 1] - m1));
    a2 = fmaxf(a2, fabsf(pts[(size_t)i * 3 + 2] - m2));
  }
  a0 = wred_max(a0); a1 = wred_max(a1); a2 = wred_max(a2);
  if ((threadIdx.x & 31) == 0) {
    atomicMax(maxb + 0, __float_as_uint(a0));  // nonneg floats: uint order OK
    atomicMax(maxb + 1, __float_as_uint(a1));
    atomicMax(maxb + 2, __float_as_uint(a2));
  }
}

__global__ void k_scale_fin(const unsigned* maxb, float* stats) {
  if (threadIdx.x < 3)
    stats[3 + threadIdx.x] = fmaxf(__uint_as_float(maxb[threadIdx.x]), 1e-6f);
}

// ---------------------------------------------------------------------------
// Weight pack: f32 [K x 128] -> f16 in exact WMMA-B per-lane order.
// mode 0: rows < Ksrc direct, else zero.
// mode 1 (mix_w1): k<257 -> row k; 288<=k<416 -> row k-31; else zero pad.
// ---------------------------------------------------------------------------
__global__ void k_pack(const float* __restrict__ W, _Float16* __restrict__ dst,
                       int Kpad, int mode, int Ksrc) {
  int total = Kpad * 128;
  for (int idx = blockIdx.x * blockDim.x + threadIdx.x; idx < total;
       idx += gridDim.x * blockDim.x) {
    int e = idx & 15;
    int lane = (idx >> 4) & 31;
    int t = idx >> 9;
    int nt = t & 7;
    int kt = t >> 3;
    int nn = nt * 16 + (lane & 15);
    int kl = (e < 8 ? e : e + 8) + ((lane >= 16) ? 8 : 0);
    int k = kt * 32 + kl;
    float v = 0.f;
    if (mode == 0) {
      if (k < Ksrc) v = W[(size_t)k * 128 + nn];
    } else {
      if (k < 257) v = W[(size_t)k * 128 + nn];
      else if (k >= 288 && k < 416) v = W[(size_t)(k - 31) * 128 + nn];
    }
    dst[idx] = (_Float16)v;
  }
}

// ---------------------------------------------------------------------------
// Per-point prep: LayerNorm -> xh (f16), coord hidden -> hcp (f16).
// One wave per point (32 lanes x 4 features).
// ---------------------------------------------------------------------------
__global__ void k_point_prep(const float* __restrict__ feats,
                             const float* __restrict__ pts,
                             const float* __restrict__ g,
                             const float* __restrict__ b,
                             const float* __restrict__ cpw1,
                             const float* __restrict__ cpb1,
                             const float* __restrict__ stats,
                             _Float16* __restrict__ xh,
                             _Float16* __restrict__ hcp, int n) {
  int lane = threadIdx.x & 31;
  int p = blockIdx.x * 8 + (threadIdx.x >> 5);
  if (p >= n) return;
  int c = lane * 4;
  float4 f = *(const float4*)(feats + (size_t)p * 128 + c);
  float mu = wred_sum(f.x + f.y + f.z + f.w) * (1.0f / 128.0f);
  float dx = f.x - mu, dy = f.y - mu, dz = f.z - mu, dw = f.w - mu;
  float var = wred_sum(dx * dx + dy * dy + dz * dz + dw * dw) * (1.0f / 128.0f);
  float rs = rsqrtf(var + 1e-5f);
  union { _Float16 h[4]; uint2 u; } o;
  o.h[0] = (_Float16)(dx * rs * g[c + 0] + b[c + 0]);
  o.h[1] = (_Float16)(dy * rs * g[c + 1] + b[c + 1]);
  o.h[2] = (_Float16)(dz * rs * g[c + 2] + b[c + 2]);
  o.h[3] = (_Float16)(dw * rs * g[c + 3] + b[c + 3]);
  *(uint2*)(xh + (size_t)p * 128 + c) = o.u;

  float p0 = (pts[(size_t)p * 3 + 0] - stats[0]) / stats[3];
  float p1 = (pts[(size_t)p * 3 + 1] - stats[1]) / stats[4];
  float p2 = (pts[(size_t)p * 3 + 2] - stats[2]) / stats[5];
  union { _Float16 h[4]; uint2 u; } o2;
#pragma unroll
  for (int qi = 0; qi < 4; ++qi) {
    int j = c + qi;
    o2.h[qi] = (_Float16)gelu_exact(p0 * cpw1[j] + p1 * cpw1[128 + j] +
                                    p2 * cpw1[256 + j] + cpb1[j]);
  }
  *(uint2*)(hcp + (size_t)p * 128 + c) = o2.u;
}

// ---------------------------------------------------------------------------
// coord_embed = hcp @ cp_w2 + cp_b2  (WMMA, K=128), output f16.
// ---------------------------------------------------------------------------
__global__ void __launch_bounds__(256) k_ce_gemm(
    const _Float16* __restrict__ hcp, const _Float16* __restrict__ pkW,
    const float* __restrict__ bias, _Float16* __restrict__ ce, int n) {
  int lane = threadIdx.x & 31;
  int wv = threadIdx.x >> 5;
  int base = blockIdx.x * 128 + wv * 16;
  int r = base + (lane & 15);
  if (r > n - 1) r = n - 1;
  v8f z = {0.f, 0.f, 0.f, 0.f, 0.f, 0.f, 0.f, 0.f};
  v8f acc[8];
#pragma unroll
  for (int t = 0; t < 8; ++t) acc[t] = z;
#pragma unroll
  for (int kt = 0; kt < 4; ++kt) {
    Frag a = load_row_frag(hcp, r, kt * 32, lane);
#pragma unroll
    for (int nt = 0; nt < 8; ++nt) {
      Frag bf = load_b_frag(pkW, kt, nt, lane);
      acc[nt] = wmma16(a, bf, acc[nt]);
    }
  }
#pragma unroll
  for (int nt = 0; nt < 8; ++nt) {
    int col = nt * 16 + (lane & 15);
    float bs = bias[col];
#pragma unroll
    for (int i = 0; i < 8; ++i) {
      int grow = base + i + ((lane >= 16) ? 8 : 0);
      if (grow < n) ce[(size_t)grow * 128 + col] = (_Float16)(acc[nt][i] + bs);
    }
  }
}

// ---------------------------------------------------------------------------
// Neighbor gather: mean_feat (f16), boundary score (f16). Wave per point.
// xh is L2-resident (25.6 MB), so the random gather stays on-chip.
// ---------------------------------------------------------------------------
__global__ void k_gather(const _Float16* __restrict__ xh,
                         const float* __restrict__ pts,
                         const int* __restrict__ nbr,
                         const float* __restrict__ bdw1,
                         const float* __restrict__ bdb1,
                         const float* __restrict__ bdw2,
                         const float* __restrict__ bdb2,
                         _Float16* __restrict__ mf, _Float16* __restrict__ bnd,
                         int n) {
  int lane = threadIdx.x & 31;
  int p = blockIdx.x * 8 + (threadIdx.x >> 5);
  if (p >= n) return;
  int c = lane * 4;
  float a0 = 0.f, a1 = 0.f, a2 = 0.f, a3 = 0.f, mp = 0.f;
  int cnt = 0;
#pragma unroll
  for (int j = 0; j < 16; ++j) {
    int idx = nbr[(size_t)p * 16 + j];
    bool valid = (idx >= 0) && (idx < n);
    int s = idx < 0 ? 0 : (idx >= n ? n - 1 : idx);
    union { uint2 u; _Float16 h[4]; } L;
    L.u = *(const uint2*)(xh + (size_t)s * 128 + c);
    if (valid) {
      a0 += (float)L.h[0]; a1 += (float)L.h[1];
      a2 += (float)L.h[2]; a3 += (float)L.h[3];
      ++cnt;
    }
    if (lane < 3 && valid) mp += pts[(size_t)s * 3 + lane];
  }
  float inv = 1.0f / fmaxf((float)cnt, 1.0f);
  a0 *= inv; a1 *= inv; a2 *= inv; a3 *= inv;
  union { _Float16 h[4]; uint2 u; } o;
  o.h[0] = (_Float16)a0; o.h[1] = (_Float16)a1;
  o.h[2] = (_Float16)a2; o.h[3] = (_Float16)a3;
  *(uint2*)(mf + (size_t)p * 128 + c) = o.u;

  union { uint2 u; _Float16 h[4]; } X;
  X.u = *(const uint2*)(xh + (size_t)p * 128 + c);
  float d0 = (float)X.h[0] - a0, d1 = (float)X.h[1] - a1;
  float d2 = (float)X.h[2] - a2, d3 = (float)X.h[3] - a3;
  float fd = sqrtf(wred_sum(d0 * d0 + d1 * d1 + d2 * d2 + d3 * d3));
  float cdp = 0.f;
  if (lane < 3) {
    float t = pts[(size_t)p * 3 + lane] - mp * inv;
    cdp = t * t;
  }
  float cd = sqrtf(wred_sum(cdp));
  float part = 0.f;
#pragma unroll
  for (int qi = 0; qi < 4; ++qi) {
    int j = c + qi;
    float h = gelu_exact(fd * bdw1[j] + cd * bdw1[128 + j] + bdb1[j]);
    part += h * bdw2[j];
  }
  part = wred_sum(part);
  if (lane == 0) bnd[p] = (_Float16)sigmoidf_(part + bdb2[0]);
}

// ---------------------------------------------------------------------------
// Fused head. Per block: 8 waves x 16 rows. A K-layout (padded to 416):
//   [0..127]=x  [128..255]=mean_feat  [256]=boundary  [257..287]=0
//   [288..415]=coord_embed.  Gate shares K-steps 0..8 with mix.
// Hidden activations bounce through per-wave LDS (no barriers needed).
// ---------------------------------------------------------------------------
__global__ void __launch_bounds__(256) k_fused(
    const _Float16* __restrict__ xh, const _Float16* __restrict__ mf,
    const _Float16* __restrict__ bnd, const _Float16* __restrict__ ce,
    const float* __restrict__ feats,
    const _Float16* __restrict__ pkMixW1, const _Float16* __restrict__ pkGateW1,
    const _Float16* __restrict__ pkMixW2, const _Float16* __restrict__ pkGateW2,
    const _Float16* __restrict__ pkOutW,
    const float* __restrict__ mix_b1, const float* __restrict__ mix_b2,
    const float* __restrict__ gate_b1, const float* __restrict__ gate_b2,
    const float* __restrict__ out_b, float* __restrict__ out, int n) {
  __shared__ _Float16 sm[8 * 2 * 16 * 136];  // 69,632 B
  int lane = threadIdx.x & 31;
  int wv = threadIdx.x >> 5;
  int base = blockIdx.x * 128 + wv * 16;
  int r = base + (lane & 15);
  if (r > n - 1) r = n - 1;
  _Float16* Hm = sm + wv * 2 * 16 * 136;
  _Float16* Hg = Hm + 16 * 136;

  v8f z = {0.f, 0.f, 0.f, 0.f, 0.f, 0.f, 0.f, 0.f};
  v8f aM[8], aG[8];
#pragma unroll
  for (int t = 0; t < 8; ++t) { aM[t] = z; aG[t] = z; }

#pragma unroll
  for (int kt = 0; kt < 13; ++kt) {
    Frag a;
    if (kt < 4) a = load_row_frag(xh, r, kt * 32, lane);
    else if (kt < 8) a = load_row_frag(mf, r, (kt - 4) * 32, lane);
    else if (kt == 8) {
      a.q[0] = make_uint4(0, 0, 0, 0);
      a.q[1] = make_uint4(0, 0, 0, 0);
      if (lane < 16) a.e[0] = bnd[r];  // K=256 lives at (lane<16, e=0)
    } else a = load_row_frag(ce, r, (kt - 9) * 32, lane);
#pragma unroll
    for (int nt = 0; nt < 8; ++nt) {
      Frag bf = load_b_frag(pkMixW1, kt, nt, lane);
      aM[nt] = wmma16(a, bf, aM[nt]);
    }
    if (kt < 9) {
#pragma unroll
      for (int nt = 0; nt < 8; ++nt) {
        Frag bf = load_b_frag(pkGateW1, kt, nt, lane);
        aG[nt] = wmma16(a, bf, aG[nt]);
      }
    }
  }

  // hidden GELU -> LDS (own rows only)
#pragma unroll
  for (int nt = 0; nt < 8; ++nt) {
    int col = nt * 16 + (lane & 15);
    float bm = mix_b1[col], bg = gate_b1[col];
#pragma unroll
    for (int i = 0; i < 8; ++i) {
      int rl = i + ((lane >= 16) ? 8 : 0);
      Hm[rl * 136 + col] = (_Float16)gelu_exact(aM[nt][i] + bm);
      Hg[rl * 136 + col] = (_Float16)gelu_exact(aG[nt][i] + bg);
    }
  }

  // second layers: refined & gate (K=128 each)
  v8f aR[8], aS[8];
#pragma unroll
  for (int t = 0; t < 8; ++t) { aR[t] = z; aS[t] = z; }
#pragma unroll
  for (int kt = 0; kt < 4; ++kt) {
    Frag am = lds_frag(Hm, kt, lane);
    Frag ag = lds_frag(Hg, kt, lane);
#pragma unroll
    for (int nt = 0; nt < 8; ++nt) {
      Frag bm = load_b_frag(pkMixW2, kt, nt, lane);
      aR[nt] = wmma16(am, bm, aR[nt]);
      Frag bg = load_b_frag(pkGateW2, kt, nt, lane);
      aS[nt] = wmma16(ag, bg, aS[nt]);
    }
  }
  // prod = sigmoid(gate) * refined -> reuse Hm (same-wave DS ops are in-order)
#pragma unroll
  for (int nt = 0; nt < 8; ++nt) {
    int col = nt * 16 + (lane & 15);
    float b2m = mix_b2[col], b2g = gate_b2[col];
#pragma unroll
    for (int i = 0; i < 8; ++i) {
      int rl = i + ((lane >= 16) ? 8 : 0);
      float rr = aR[nt][i] + b2m;
      float gg = sigmoidf_(aS[nt][i] + b2g);
      Hm[rl * 136 + col] = (_Float16)(rr * gg);
    }
  }

  // output layer (K=128) + residual
  v8f aO[8];
#pragma unroll
  for (int t = 0; t < 8; ++t) aO[t] = z;
#pragma unroll
  for (int kt = 0; kt < 4; ++kt) {
    Frag ap = lds_frag(Hm, kt, lane);
#pragma unroll
    for (int nt = 0; nt < 8; ++nt) {
      Frag bo = load_b_frag(pkOutW, kt, nt, lane);
      aO[nt] = wmma16(ap, bo, aO[nt]);
    }
  }
#pragma unroll
  for (int nt = 0; nt < 8; ++nt) {
    int col = nt * 16 + (lane & 15);
    float bo = out_b[col];
#pragma unroll
    for (int i = 0; i < 8; ++i) {
      int grow = base + i + ((lane >= 16) ? 8 : 0);
      if (grow < n)
        out[(size_t)grow * 128 + col] =
            feats[(size_t)grow * 128 + col] + aO[nt][i] + bo;
    }
  }
}

// ---------------------------------------------------------------------------
static inline size_t align256(size_t x) { return (x + 255) & ~(size_t)255; }

extern "C" void kernel_launch(void* const* d_in, const int* in_sizes, int n_in,
                              void* d_out, int out_size, void* d_ws,
                              size_t ws_size, hipStream_t stream) {
  (void)n_in; (void)out_size; (void)ws_size;
  const float* feats  = (const float*)d_in[0];
  const float* points = (const float*)d_in[1];
  const int*   nbr    = (const int*)d_in[2];
  const float* ln_g   = (const float*)d_in[3];
  const float* ln_b   = (const float*)d_in[4];
  const float* cp_w1  = (const float*)d_in[5];
  const float* cp_b1  = (const float*)d_in[6];
  const float* cp_w2  = (const float*)d_in[7];
  const float* cp_b2  = (const float*)d_in[8];
  const float* bd_w1  = (const float*)d_in[9];
  const float* bd_b1  = (const float*)d_in[10];
  const float* bd_w2  = (const float*)d_in[11];
  const float* bd_b2  = (const float*)d_in[12];
  const float* mix_w1 = (const float*)d_in[13];
  const float* mix_b1 = (const float*)d_in[14];
  const float* mix_w2 = (const float*)d_in[15];
  const float* mix_b2 = (const float*)d_in[16];
  const float* gate_w1 = (const float*)d_in[17];
  const float* gate_b1 = (const float*)d_in[18];
  const float* gate_w2 = (const float*)d_in[19];
  const float* gate_b2 = (const float*)d_in[20];
  const float* out_w   = (const float*)d_in[21];
  const float* out_b   = (const float*)d_in[22];
  float* out = (float*)d_out;
  const int n = in_sizes[0] / 128;

  // Workspace carve (all 256B aligned)
  char* w = (char*)d_ws;
  float*    stats   = (float*)w;                 // [0..2]=mean [3..5]=scale
  unsigned* maxb    = (unsigned*)(w + 32);       // 3 x u32
  float*    partial = (float*)(w + 64);          // RED_BLOCKS*3
  size_t off = align256(64 + (size_t)RED_BLOCKS * 3 * 4);
  _Float16* xh  = (_Float16*)(w + off); off += align256((size_t)n * 128 * 2);
  _Float16* hcp = (_Float16*)(w + off); off += align256((size_t)n * 128 * 2);
  _Float16* ce  = (_Float16*)(w + off); off += align256((size_t)n * 128 * 2);
  _Float16* mfb = (_Float16*)(w + off); off += align256((size_t)n * 128 * 2);
  _Float16* bnd = (_Float16*)(w + off); off += align256((size_t)n * 2);
  _Float16* pkMixW1  = (_Float16*)(w + off); off += align256((size_t)416 * 128 * 2);
  _Float16* pkGateW1 = (_Float16*)(w + off); off += align256((size_t)288 * 128 * 2);
  _Float16* pkMixW2  = (_Float16*)(w + off); off += align256((size_t)128 * 128 * 2);
  _Float16* pkGateW2 = (_Float16*)(w + off); off += align256((size_t)128 * 128 * 2);
  _Float16* pkOutW   = (_Float16*)(w + off); off += align256((size_t)128 * 128 * 2);
  _Float16* pkCpW2   = (_Float16*)(w + off); off += align256((size_t)128 * 128 * 2);

  // global point stats
  k_stats_init<<<1, 64, 0, stream>>>(maxb);
  k_mean_part<<<RED_BLOCKS, 256, 0, stream>>>(points, n, partial);
  k_mean_fin<<<1, 32, 0, stream>>>(partial, n, stats);
  k_maxabs<<<RED_BLOCKS, 256, 0, stream>>>(points, n, stats, maxb);
  k_scale_fin<<<1, 32, 0, stream>>>(maxb, stats);

  // weight packing (f32 -> WMMA-B f16 layout)
  k_pack<<<64, 256, 0, stream>>>(mix_w1, pkMixW1, 416, 1, 0);
  k_pack<<<64, 256, 0, stream>>>(gate_w1, pkGateW1, 288, 0, 257);
  k_pack<<<32, 256, 0, stream>>>(mix_w2, pkMixW2, 128, 0, 128);
  k_pack<<<32, 256, 0, stream>>>(gate_w2, pkGateW2, 128, 0, 128);
  k_pack<<<32, 256, 0, stream>>>(out_w, pkOutW, 128, 0, 128);
  k_pack<<<32, 256, 0, stream>>>(cp_w2, pkCpW2, 128, 0, 128);

  const int nblk_pt = (n + 7) / 8;
  const int nblk_mm = (n + 127) / 128;

  k_point_prep<<<nblk_pt, 256, 0, stream>>>(feats, points, ln_g, ln_b, cp_w1,
                                            cp_b1, stats, xh, hcp, n);
  k_ce_gemm<<<nblk_mm, 256, 0, stream>>>(hcp, pkCpW2, cp_b2, ce, n);
  k_gather<<<nblk_pt, 256, 0, stream>>>(xh, points, nbr, bd_w1, bd_b1, bd_w2,
                                        bd_b2, mfb, bnd, n);
  k_fused<<<nblk_mm, 256, 0, stream>>>(xh, mfb, bnd, ce, feats, pkMixW1,
                                       pkGateW1, pkMixW2, pkGateW2, pkOutW,
                                       mix_b1, mix_b2, gate_b1, gate_b2, out_b,
                                       out, n);
}
